// DecNet_19705309954357
// MI455X (gfx1250) — matrix operand order repaired and checked by reading, not compiled, not run on hardware
//
#include <hip/hip_runtime.h>
#include <cstddef>
#include <cstdint>

typedef __bf16 bf16;
typedef __attribute__((ext_vector_type(16))) __bf16 v16bf;
typedef __attribute__((ext_vector_type(16))) float  v16f;
typedef __attribute__((ext_vector_type(8)))  float  v8f;

#define PI_F 3.14159265358979f

__device__ __forceinline__ float silu_f(float v) { return v / (1.0f + __expf(-v)); }

// ---------------------------------------------------------------------------
// WMMA GEMM:  D[M,N] = act( A[M,K](f32) * Wb[K,N](bf16) + bias (+D) )
// Compile-time K (multiple of 32).  M multiple of 16, N multiple of 64.
// Each wave computes a 16x64 strip: one A fragment, four B fragments,
// 4 independent v_wmma_f32_16x16x32_bf16 per k-step.
// act: 0=none, 1=silu, 2=softplus(x)-ln2.
// ---------------------------------------------------------------------------
template <int K>
__global__ void gemm_bf16_wmma(const float* __restrict__ A,
                               const bf16*  __restrict__ Wb,
                               const float* __restrict__ bias,
                               float* __restrict__ D,
                               int M, int N, int act, int accumulate)
{
    int wave = blockIdx.x * (blockDim.x >> 5) + (threadIdx.x >> 5);
    int lane = threadIdx.x & 31;
    int mtiles  = M >> 4;
    int ngroups = N >> 6;                  // 64-wide N strips
    if (wave >= mtiles * ngroups) return;
    int mt = wave / ngroups, ng = wave - mt * ngroups;
    int m0 = mt << 4, n0 = ng << 6;
    int lrow  = lane & 15;
    int lhalf = lane >> 4;

    // A: row m0+lrow; element i -> K index ((i&8)<<1) + lhalf*8 + (i&7)
    const float* ap = A + (size_t)(m0 + lrow) * K + (lhalf << 3);
    // B: lane holds N col lrow (+16t); element i -> K index ((i&8)<<1) + lhalf*8 + (i&7)
    const bf16* bp0 = Wb + (size_t)(lhalf << 3) * N + n0 + lrow;

    v8f acc0 = {}, acc1 = {}, acc2 = {}, acc3 = {};
    #pragma unroll
    for (int k0 = 0; k0 < K; k0 += 32) {
        if (k0 + 32 < K)                    // prefetch next B k-row (128B/lane)
            __builtin_prefetch(bp0 + (size_t)(k0 + 32) * N, 0, 1);
        v8f lo = *(const v8f*)(ap + k0);        // K = k0+lhalf*8 + 0..7
        v8f hi = *(const v8f*)(ap + k0 + 16);   // K = k0+16+lhalf*8 + 0..7
        v16f af = __builtin_shufflevector(lo, hi, 0, 1, 2, 3, 4, 5, 6, 7,
                                                  8, 9, 10, 11, 12, 13, 14, 15);
        v16bf a = __builtin_convertvector(af, v16bf);
        const bf16* bp = bp0 + (size_t)k0 * N;  // strided K gather per N col
        v16bf b0, b1, b2, b3;
        #pragma unroll
        for (int i = 0; i < 16; ++i) {
            size_t ko = (size_t)(((i & 8) << 1) + (i & 7)) * N;
            b0[i] = bp[ko];
            b1[i] = bp[ko + 16];
            b2[i] = bp[ko + 32];
            b3[i] = bp[ko + 48];
        }
        acc0 = __builtin_amdgcn_wmma_f32_16x16x32_bf16(false, a, false, b0, (short)0, acc0, false, false);
        acc1 = __builtin_amdgcn_wmma_f32_16x16x32_bf16(false, a, false, b1, (short)0, acc1, false, false);
        acc2 = __builtin_amdgcn_wmma_f32_16x16x32_bf16(false, a, false, b2, (short)0, acc2, false, false);
        acc3 = __builtin_amdgcn_wmma_f32_16x16x32_bf16(false, a, false, b3, (short)0, acc3, false, false);
    }

    int mbase = m0 + (lhalf << 3);         // C/D: M = r + 8*(lane>=16)
    v8f accs[4] = { acc0, acc1, acc2, acc3 };
    #pragma unroll
    for (int t = 0; t < 4; ++t) {
        int nn = n0 + (t << 4) + lrow;     // C/D: N = lane&15
        float bv = bias ? bias[nn] : 0.0f;
        float* dp = D + (size_t)mbase * N + nn;
        // Batched read-modify path: uniform branch around a clause of 8 loads
        // (keeps the accumulate GEMMs from serializing 8 load-waits).
        float prev[8];
        if (accumulate) {
            #pragma unroll
            for (int r = 0; r < 8; ++r) prev[r] = dp[(size_t)r * N];
        }
        #pragma unroll
        for (int r = 0; r < 8; ++r) {
            float v = accs[t][r] + bv;
            if (accumulate) v += prev[r];
            if (act == 1)      v = silu_f(v);
            else if (act == 2) { v = (v > 20.0f) ? v : log1pf(__expf(v));
                                 v -= 0.69314718056f; }
            dp[(size_t)r * N] = v;
        }
    }
}

// fp32 -> bf16 weight conversion (layout preserved, row-major K x N)
__global__ void conv_bf16_kernel(const float* __restrict__ W,
                                 bf16* __restrict__ Wb, int total)
{
    int t = blockIdx.x * blockDim.x + threadIdx.x;
    if (t < total) Wb[t] = (bf16)W[t];
}

// fp32 (K0 x N) -> bf16 (K x N) with zero-padded rows K0..K-1
__global__ void conv_pad_bf16_kernel(const float* __restrict__ W,
                                     bf16* __restrict__ Wb, int K0, int N, int K)
{
    int t = blockIdx.x * blockDim.x + threadIdx.x;
    if (t >= K * N) return;
    int k = t / N, n = t - k * N;
    Wb[t] = (k < K0) ? (bf16)W[(size_t)k * N + n] : (bf16)0.0f;
}

// fp32 (M x K0) -> fp32 (M x K) with zero-padded cols K0..K-1
__global__ void pad_rows_kernel(const float* __restrict__ src,
                                float* __restrict__ dst, int M, int K0, int K)
{
    int t = blockIdx.x * blockDim.x + threadIdx.x;
    if (t >= M * K) return;
    int r = t / K, c = t - r * K;
    dst[t] = (c < K0) ? src[(size_t)r * K0 + c] : 0.0f;
}

// Per-edge geometry: Bessel RBF (8 of 32, zero-padded), sph-harm (9), cutoff.
__global__ void edge_geom_kernel(const float* __restrict__ pos,
                                 const int* __restrict__ ei, int E,
                                 float* __restrict__ rbf32,
                                 float* __restrict__ Yb,
                                 float* __restrict__ cut)
{
    int e = blockIdx.x * blockDim.x + threadIdx.x;
    if (e >= E) return;
    int i = ei[e], j = ei[E + e];
    float dx = pos[3*j+0] - pos[3*i+0];
    float dy = pos[3*j+1] - pos[3*i+1];
    float dz = pos[3*j+2] - pos[3*i+2];
    float d  = sqrtf(dx*dx + dy*dy + dz*dz + 1e-12f);
    float inv = 1.0f / d;
    float x  = d * 0.2f;                       // d / R_MAX
    float x3 = x*x*x, x6 = x3*x3, x7 = x6*x, x8 = x7*x;
    float env = (x < 1.0f) ? (1.0f - 28.0f*x6 + 48.0f*x7 - 21.0f*x8) : 0.0f;
    float se  = 0.632455532f * inv * env;      // sqrt(2/R_MAX)
    float* rp = rbf32 + (size_t)e * 32;
    #pragma unroll
    for (int n = 1; n <= 8; ++n)
        rp[n-1] = se * __sinf((float)n * PI_F * x);
    #pragma unroll
    for (int n = 8; n < 32; ++n) rp[n] = 0.0f;
    float ux = dx*inv, uy = dy*inv, uz = dz*inv;
    const float s3 = 1.7320508f, s5 = 2.236068f, s15 = 3.8729833f;
    float* yp = Yb + (size_t)e*9;
    yp[0] = 1.0f;        yp[1] = s3*ux;       yp[2] = s3*uy;
    yp[3] = s3*uz;       yp[4] = s15*ux*uy;   yp[5] = s15*uy*uz;
    yp[6] = 0.5f*s5*(3.0f*uz*uz - 1.0f);
    yp[7] = s15*ux*uz;   yp[8] = 0.5f*s15*(ux*ux - uy*uy);
    float xc = fminf(fmaxf(x, 0.0f), 1.0f);
    cut[e] = 0.5f * (__cosf(PI_F * xc) + 1.0f);
}

// msg = hj * w[lmap[m]] * Y[m] / 3, scattered into agg[m][idx_i][c] (L2 atomics)
__global__ void message_scatter_kernel(const int* __restrict__ ei, int E, int Nn,
                                       int e0, int ecnt,
                                       const float* __restrict__ rout,
                                       const float* __restrict__ feats0,
                                       const float* __restrict__ Yb,
                                       float* __restrict__ agg)
{
    int t = blockIdx.x * blockDim.x + threadIdx.x;
    if (t >= ecnt * 128) return;
    int el = t >> 7, c = t & 127;
    int e  = e0 + el;
    int ni = ei[e], nj = ei[E + e];
    float hj = feats0[(size_t)nj*128 + c];
    const float* wp = rout + (size_t)el*384 + c*3;
    float w[3] = { wp[0], wp[1], wp[2] };
    const float* y = Yb + (size_t)e*9;
    size_t slab = (size_t)Nn * 128;
    float* outp = agg + (size_t)ni*128 + c;
    const float inv3 = 1.0f / 3.0f;
    #pragma unroll
    for (int m = 0; m < 9; ++m) {
        int l = (m == 0) ? 0 : ((m < 4) ? 1 : 2);
        atomicAdd(outp + (size_t)m * slab, hj * w[l] * y[m] * inv3);
    }
}

__global__ void scalar_scatter_kernel(const int* __restrict__ ei, int E,
                                      int e0, int ecnt,
                                      const float* __restrict__ node_sca,
                                      const float* __restrict__ e_emb,
                                      const float* __restrict__ cut,
                                      float* __restrict__ agg_s)
{
    int t = blockIdx.x * blockDim.x + threadIdx.x;
    if (t >= ecnt * 128) return;
    int el = t >> 7, c = t & 127;
    int e  = e0 + el;
    int ni = ei[e], nj = ei[E + e];
    float v = node_sca[(size_t)nj*128 + c] * e_emb[(size_t)el*128 + c] * cut[e];
    atomicAdd(&agg_s[(size_t)ni*128 + c], v);
}

__global__ void concat_kernel(const float* __restrict__ a,
                              const float* __restrict__ b,
                              float* __restrict__ hcat, int Nn)
{
    int t = blockIdx.x * blockDim.x + threadIdx.x;
    if (t >= Nn * 256) return;
    int n = t >> 8, c = t & 255;
    hcat[t] = (c < 128) ? a[(size_t)n*128 + c] : b[(size_t)n*128 + (c - 128)];
}

// Final N=1 projection (VALU dot; too thin for WMMA)
__global__ void ro4_kernel(const float* __restrict__ h3,
                           const float* __restrict__ W,
                           const float* __restrict__ b,
                           float* __restrict__ out, int Nn)
{
    int n = blockIdx.x * blockDim.x + threadIdx.x;
    if (n >= Nn) return;
    float s = b[0];
    #pragma unroll
    for (int k = 0; k < 64; ++k) s += h3[(size_t)n*64 + k] * W[k];
    out[n] = s;
}

// ---------------------------------------------------------------------------
extern "C" void kernel_launch(void* const* d_in, const int* in_sizes, int n_in,
                              void* d_out, int out_size, void* d_ws, size_t ws_size,
                              hipStream_t stream)
{
    const float* pos     = (const float*)d_in[0];
    const float* Z       = (const float*)d_in[1];
    const int*   ei      = (const int*)  d_in[2];
    const float* W_embed = (const float*)d_in[3];
    const float* W_rad1  = (const float*)d_in[4];
    const float* W_rad2  = (const float*)d_in[5];
    const float* W_rad3  = (const float*)d_in[6];
    const float* W_rad4  = (const float*)d_in[7];
    const float* W_mix   = (const float*)d_in[8];
    const float* W_skip  = (const float*)d_in[9];
    const float* W_v2s   = (const float*)d_in[10];
    const float* b_v2s   = (const float*)d_in[11];
    const float* W_edge  = (const float*)d_in[12];
    const float* b_edge  = (const float*)d_in[13];
    const float* W_hil   = (const float*)d_in[14];
    const float* b_hil   = (const float*)d_in[15];
    const float* W_ro1   = (const float*)d_in[16];
    const float* b_ro1   = (const float*)d_in[17];
    const float* W_ro2   = (const float*)d_in[18];
    const float* b_ro2   = (const float*)d_in[19];
    const float* W_ro3   = (const float*)d_in[20];
    const float* b_ro3   = (const float*)d_in[21];
    const float* W_ro4   = (const float*)d_in[22];
    const float* b_ro4   = (const float*)d_in[23];

    const int N = in_sizes[0] / 3;
    const int E = in_sizes[2] / 2;
    const int C = 128;
    const int ECH = 20000;              // edge chunk (multiple of 16)

    // ---- workspace carve ----
    size_t cur = 0;
    auto carve = [&](size_t bytes) -> void* {
        void* p = (char*)d_ws + cur;
        cur += (bytes + 255) & ~(size_t)255;
        return p;
    };
    float* rbf32  = (float*)carve((size_t)E * 32 * 4);   // zero-padded K=32
    float* Yb     = (float*)carve((size_t)E * 9 * 4);
    float* cut    = (float*)carve((size_t)E * 4);
    float* Zpad   = (float*)carve((size_t)N * 32 * 4);   // zero-padded K=32
    float* featsA = (float*)carve((size_t)9 * N * C * 4);
    float* featsB = (float*)carve((size_t)9 * N * C * 4);
    float* agg    = (float*)carve((size_t)9 * N * C * 4);
    float* h_a    = (float*)carve((size_t)ECH * 64 * 4);
    float* h_b    = (float*)carve((size_t)ECH * 64 * 4);
    float* rout   = (float*)carve((size_t)ECH * 384 * 4);
    float* e_emb  = (float*)carve((size_t)ECH * C * 4);
    float* node_sca    = (float*)carve((size_t)N * C * 4);
    float* agg_s       = (float*)carve((size_t)N * C * 4);
    float* node_scalar = (float*)carve((size_t)N * C * 4);
    float* hcat = (float*)carve((size_t)N * 256 * 4);
    float* h1   = (float*)carve((size_t)N * 256 * 4);
    float* h2   = (float*)carve((size_t)N * 128 * 4);
    float* h3   = (float*)carve((size_t)N * 64 * 4);
    bf16* embedB = (bf16*)carve(32 * 128 * 2);           // padded K=32
    bf16* rad1B  = (bf16*)carve(3 * 32 * 64 * 2);        // padded K=32
    bf16* rad2B  = (bf16*)carve(3 * 64 * 64 * 2);
    bf16* rad3B  = (bf16*)carve(3 * 64 * 64 * 2);
    bf16* rad4B  = (bf16*)carve(3 * 64 * 384 * 2);
    bf16* mixB   = (bf16*)carve(9 * 128 * 128 * 2);
    bf16* skipB  = (bf16*)carve(9 * 128 * 128 * 2);
    bf16* v2sB   = (bf16*)carve(128 * 128 * 2);
    bf16* edgeB  = (bf16*)carve(32 * 128 * 2);           // padded K=32
    bf16* hilB   = (bf16*)carve(128 * 128 * 2);
    bf16* ro1B   = (bf16*)carve(256 * 256 * 2);
    bf16* ro2B   = (bf16*)carve(256 * 128 * 2);
    bf16* ro3B   = (bf16*)carve(128 * 64 * 2);

    auto conv = [&](const float* W, bf16* Wb, int total) {
        conv_bf16_kernel<<<(total + 255) / 256, 256, 0, stream>>>(W, Wb, total);
    };
    auto conv_pad = [&](const float* W, bf16* Wb, int K0, int Nc, int K) {
        int total = K * Nc;
        conv_pad_bf16_kernel<<<(total + 255) / 256, 256, 0, stream>>>(W, Wb, K0, Nc, K);
    };
    auto gemm = [&](const float* A, const bf16* Wb, const float* bias, float* D,
                    int M, int Nc, int K, int act, int accum) {
        int tiles = (M / 16) * (Nc / 64);             // 16x64 strip per wave
        int blocks = (tiles + 7) / 8;                 // 8 waves / 256-thread block
        switch (K) {
        case 32:  gemm_bf16_wmma<32><<<blocks, 256, 0, stream>>>(A, Wb, bias, D, M, Nc, act, accum); break;
        case 64:  gemm_bf16_wmma<64><<<blocks, 256, 0, stream>>>(A, Wb, bias, D, M, Nc, act, accum); break;
        case 128: gemm_bf16_wmma<128><<<blocks, 256, 0, stream>>>(A, Wb, bias, D, M, Nc, act, accum); break;
        case 256: gemm_bf16_wmma<256><<<blocks, 256, 0, stream>>>(A, Wb, bias, D, M, Nc, act, accum); break;
        }
    };

    // ---- weight prep (bf16; ragged K padded to 32) ----
    conv_pad(W_embed, embedB, 10, 128, 32);
    for (int k = 0; k < 3; ++k)
        conv_pad(W_rad1 + (size_t)k * 8 * 64, rad1B + (size_t)k * 32 * 64, 8, 64, 32);
    conv(W_rad2, rad2B, 3 * 64 * 64);
    conv(W_rad3, rad3B, 3 * 64 * 64);
    conv(W_rad4, rad4B, 3 * 64 * 384);
    conv(W_mix,  mixB,  9 * 128 * 128);
    conv(W_skip, skipB, 9 * 128 * 128);
    conv(W_v2s,  v2sB,  128 * 128);
    conv_pad(W_edge, edgeB, 8, 128, 32);
    conv(W_hil,  hilB,  128 * 128);
    conv(W_ro1,  ro1B,  256 * 256);
    conv(W_ro2,  ro2B,  256 * 128);
    conv(W_ro3,  ro3B,  128 * 64);

    // ---- padded activations ----
    pad_rows_kernel<<<(N * 32 + 255) / 256, 256, 0, stream>>>(Z, Zpad, N, 10, 32);

    // ---- edge geometry ----
    edge_geom_kernel<<<(E + 255) / 256, 256, 0, stream>>>(pos, ei, E, rbf32, Yb, cut);

    // ---- node embedding: feats[m=0] = Z @ W_embed, other channels zero ----
    hipMemsetAsync(featsA, 0, (size_t)9 * N * C * 4, stream);
    gemm(Zpad, embedB, nullptr, featsA, N, C, 32, 0, 0);

    const int lmap[9] = {0, 1, 1, 1, 2, 2, 2, 2, 2};
    const size_t NC = (size_t)N * C;

    float* fcur = featsA;
    float* fnew = featsB;
    for (int k = 0; k < 3; ++k) {
        hipMemsetAsync(agg, 0, (size_t)9 * N * C * 4, stream);
        for (int e0 = 0; e0 < E; e0 += ECH) {
            int ecnt = (E - e0 < ECH) ? (E - e0) : ECH;
            gemm(rbf32 + (size_t)e0 * 32, rad1B + (size_t)k * 32 * 64, nullptr, h_a, ecnt, 64, 32, 1, 0);
            gemm(h_a, rad2B + (size_t)k * 64 * 64,  nullptr, h_b,  ecnt, 64,  64, 1, 0);
            gemm(h_b, rad3B + (size_t)k * 64 * 64,  nullptr, h_a,  ecnt, 64,  64, 1, 0);
            gemm(h_a, rad4B + (size_t)k * 64 * 384, nullptr, rout, ecnt, 384, 64, 0, 0);
            int nthr = ecnt * 128;
            message_scatter_kernel<<<(nthr + 255) / 256, 256, 0, stream>>>(
                ei, E, N, e0, ecnt, rout, fcur, Yb, agg);
        }
        for (int m = 0; m < 9; ++m) {
            const bf16* wm = mixB  + (size_t)(k * 3 + lmap[m]) * 128 * 128;
            const bf16* ws = skipB + (size_t)(k * 3 + lmap[m]) * 128 * 128;
            gemm(agg  + (size_t)m * NC, wm, nullptr, fnew + (size_t)m * NC, N, C, C, 0, 0);
            gemm(fcur + (size_t)m * NC, ws, nullptr, fnew + (size_t)m * NC, N, C, C, 0, 1);
        }
        float* t = fcur; fcur = fnew; fnew = t;
    }

    // ---- scalar head: softplus(feats0 @ W_v2s + b) - ln2 ----
    gemm(fcur, v2sB, b_v2s, node_sca, N, C, C, 2, 0);

    // ---- scalar message pass ----
    hipMemsetAsync(agg_s, 0, (size_t)N * C * 4, stream);
    for (int e0 = 0; e0 < E; e0 += ECH) {
        int ecnt = (E - e0 < ECH) ? (E - e0) : ECH;
        gemm(rbf32 + (size_t)e0 * 32, edgeB, b_edge, e_emb, ecnt, C, 32, 1, 0);
        int nthr = ecnt * 128;
        scalar_scatter_kernel<<<(nthr + 255) / 256, 256, 0, stream>>>(
            ei, E, e0, ecnt, node_sca, e_emb, cut, agg_s);
    }
    gemm(agg_s, hilB, b_hil, node_scalar, N, C, C, 1, 0);

    // ---- readout MLP ----
    concat_kernel<<<(N * 256 + 255) / 256, 256, 0, stream>>>(node_sca, node_scalar, hcat, N);
    gemm(hcat, ro1B, b_ro1, h1, N, 256, 256, 1, 0);
    gemm(h1,   ro2B, b_ro2, h2, N, 128, 256, 1, 0);
    gemm(h2,   ro3B, b_ro3, h3, N, 64, 128, 1, 0);
    ro4_kernel<<<(N + 255) / 256, 256, 0, stream>>>(h3, W_ro4, b_ro4, (float*)d_out, N);
}
